// RoPETransformer_24764781428956
// MI455X (gfx1250) — matrix operand order repaired and checked by reading, not compiled, not run on hardware
//
#include <hip/hip_runtime.h>
#include <hip/hip_bf16.h>

// ---------------- model constants ----------------
#define V_   32000
#define D_   512
#define NL_  2
#define H_   8
#define DFF_ 2048
#define B_   2
#define S_   1024
#define DK_  64          // D_/H_
#define MR_  (B_*S_)     // 2048 rows of activations

typedef __attribute__((ext_vector_type(16))) __bf16 v16bf;
typedef __attribute__((ext_vector_type(8)))  float  v8f;

// float -> bf16 bits, round-to-nearest-even
__device__ __forceinline__ unsigned short f2bf(float f) {
  union { float f; unsigned int u; } x; x.f = f;
  unsigned int u = x.u;
  u += 0x7FFFu + ((u >> 16) & 1u);
  return (unsigned short)(u >> 16);
}
__device__ __forceinline__ unsigned int pk2bf(float a, float b) {
  return (unsigned int)f2bf(a) | ((unsigned int)f2bf(b) << 16);
}

// =====================================================================
// Tiled bf16 WMMA GEMM:  C[M,N] = A[M,K] @ B (+bias) (ReLU)
//   Block: 128 threads = 4 wave32 waves. Block tile 128(M) x 64(N), K-step 64.
//   Each wave computes 32x64 -> 8 f32 accumulators, 16 WMMAs per K iter.
// BMODE: 0 = B is fp32 [K,N] (fallback), 1 = B is fp32 [N,K], 2 = B is bf16 [N,K].
// All dims used are multiples of 128 (M), 64 (N, K): no edge handling.
// =====================================================================
template<int BMODE, bool HAS_BIAS, bool RELU>
__global__ __launch_bounds__(128) void gemm_bf16_wmma(
    const float* __restrict__ A, const void* __restrict__ Bsrc,
    const float* __restrict__ bias, float* __restrict__ C,
    int Ndim, int Kdim,
    long long sAb, long long sBb, long long sCb)
{
  __shared__ __align__(16) unsigned short As[128 * 64];   // [m][k] bf16
  __shared__ __align__(16) unsigned short Bs[64 * 64];    // [n][k] bf16 (N-major)

  const int tid  = threadIdx.x;
  const int lane = tid & 31;
  const int wave = tid >> 5;
  const int z    = blockIdx.z;

  A += (long long)z * sAb;
  C += (long long)z * sCb;
  const float*          Bf = (const float*)Bsrc;
  const unsigned short* Bh = (const unsigned short*)Bsrc;
  if (BMODE == 2) Bh += (long long)z * sBb; else Bf += (long long)z * sBb;

  const int m0 = blockIdx.y * 128;
  const int n0 = blockIdx.x * 64;
  const int lda = Kdim;
  const int ldb = (BMODE == 0) ? Ndim : Kdim;
  const int ldc = Ndim;

  const int wm = wave * 32;          // wave's 32-row band
  const int lr = lane & 15;
  const int hi = lane >> 4;

  v8f acc[2][4] = {};

  for (int k0 = 0; k0 < Kdim; k0 += 64) {
    // ---- stage A tile (128x64) fp32 -> bf16, 128-bit loads, 64-bit LDS stores
    #pragma unroll
    for (int i = 0; i < 16; ++i) {
      int idx = tid + i * 128;                 // 2048 float4 chunks
      int r = idx >> 4, c = (idx & 15) << 2;
      float4 f = *(const float4*)&A[(long long)(m0 + r) * lda + k0 + c];
      unsigned long long pk = (unsigned long long)pk2bf(f.x, f.y)
                            | ((unsigned long long)pk2bf(f.z, f.w) << 32);
      *(unsigned long long*)&As[r * 64 + c] = pk;
    }
    // ---- stage B tile (64x64) as [n][k] bf16 ----
    if (BMODE == 2) {
      #pragma unroll
      for (int i = 0; i < 8; ++i) {
        int idx = tid + i * 128;               // 1024 4-elem chunks
        int n = idx >> 4, c = (idx & 15) << 2;
        uint2 pk = *(const uint2*)&Bh[(long long)(n0 + n) * ldb + k0 + c];
        *(uint2*)&Bs[n * 64 + c] = pk;
      }
    } else if (BMODE == 1) {
      #pragma unroll
      for (int i = 0; i < 8; ++i) {
        int idx = tid + i * 128;
        int n = idx >> 4, c = (idx & 15) << 2;
        float4 f = *(const float4*)&Bf[(long long)(n0 + n) * ldb + k0 + c];
        uint2 pk; pk.x = pk2bf(f.x, f.y); pk.y = pk2bf(f.z, f.w);
        *(uint2*)&Bs[n * 64 + c] = pk;
      }
    } else {                                    // fp32 [K,N] fallback
      #pragma unroll
      for (int i = 0; i < 8; ++i) {
        int idx = tid + i * 128;
        int k = idx >> 4, n4 = (idx & 15) << 2;
        float4 f = *(const float4*)&Bf[(long long)(k0 + k) * ldb + n0 + n4];
        Bs[(n4 + 0) * 64 + k] = f2bf(f.x);
        Bs[(n4 + 1) * 64 + k] = f2bf(f.y);
        Bs[(n4 + 2) * 64 + k] = f2bf(f.z);
        Bs[(n4 + 3) * 64 + k] = f2bf(f.w);
      }
    }
    __syncthreads();

    // ---- two 32-K sub-steps; fragments per CDNA5 16-bit WMMA layouts ----
    #pragma unroll
    for (int kk = 0; kk < 64; kk += 32) {
      union Frag { v16bf v; unsigned int u[8]; } a[2], b[4];
      #pragma unroll
      for (int t = 0; t < 2; ++t) {
        const unsigned short* arow = &As[(wm + t * 16 + lr) * 64 + kk];
        #pragma unroll
        for (int vv = 0; vv < 4; ++vv)
          a[t].u[vv] = *(const unsigned int*)&arow[hi * 8 + 2 * vv];
        #pragma unroll
        for (int vv = 4; vv < 8; ++vv)
          a[t].u[vv] = *(const unsigned int*)&arow[16 + hi * 8 + 2 * (vv - 4)];
      }
      #pragma unroll
      for (int t = 0; t < 4; ++t) {
        const unsigned short* brow = &Bs[(t * 16 + lr) * 64 + kk];
        #pragma unroll
        for (int vv = 0; vv < 8; ++vv)
          b[t].u[vv] = *(const unsigned int*)&brow[vv < 4 ? hi * 8 + 2 * vv : 16 + hi * 8 + 2 * (vv - 4)];
      }
      #pragma unroll
      for (int i = 0; i < 2; ++i)
        #pragma unroll
        for (int j = 0; j < 4; ++j)
          acc[i][j] = __builtin_amdgcn_wmma_f32_16x16x32_bf16(
              false, a[i].v, false, b[j].v, (short)0, acc[i][j], false, false);
    }
    __syncthreads();
  }

  // ---- epilogue (C frag: VGPR r -> M = hi*8+r, N = lr) ----
  #pragma unroll
  for (int i = 0; i < 2; ++i) {
    #pragma unroll
    for (int j = 0; j < 4; ++j) {
      int nIdx = n0 + j * 16 + lr;
      float bv = HAS_BIAS ? bias[nIdx] : 0.f;
      #pragma unroll
      for (int r = 0; r < 8; ++r) {
        int mIdx = m0 + wm + i * 16 + hi * 8 + r;
        float val = acc[i][j][r] + bv;
        if (RELU) val = fmaxf(val, 0.f);
        C[(long long)mIdx * ldc + nIdx] = val;
      }
    }
  }
}

// ======== one-time weight convert+transpose: W[K,N] f32 -> Wt[N,K] bf16 ======
__global__ __launch_bounds__(256) void conv_transpose_kernel(
    const float* __restrict__ in, unsigned short* __restrict__ out,
    int K, int N) {
  __shared__ float t[32][33];
  int n0 = blockIdx.x * 32, k0 = blockIdx.y * 32;
  int tx = threadIdx.x & 31, ty = threadIdx.x >> 5;        // 32x8
  for (int r = ty; r < 32; r += 8)
    t[r][tx] = in[(long long)(k0 + r) * N + n0 + tx];      // coalesced in n
  __syncthreads();
  for (int r = ty; r < 32; r += 8)
    out[(long long)(n0 + r) * K + k0 + tx] = f2bf(t[tx][r]); // coalesced in k
}

// ============================ elementwise ============================

__global__ void embed_kernel(const float* __restrict__ table,
                             const int* __restrict__ ids,
                             float* __restrict__ out, int nElem) {
  int i = blockIdx.x * blockDim.x + threadIdx.x;
  if (i >= nElem) return;
  int tok = i / D_, d = i % D_;
  out[i] = table[(long long)ids[tok] * D_ + d];
}

// mode 1: RoPE -> [B,H,S,DK];  mode 2: plain copy -> [B,H,DK,S] (transposed V)
__global__ void reshape_rope_kernel(const float* __restrict__ in,
                                    float* __restrict__ out, int mode) {
  int i = blockIdx.x * blockDim.x + threadIdx.x;
  const int total = B_ * H_ * S_ * (DK_ / 2);
  if (i >= total) return;
  int p = i % (DK_ / 2); int t = i / (DK_ / 2);
  int s = t % S_; t /= S_;
  int h = t % H_; int b = t / H_;
  long long inOff = (long long)(b * S_ + s) * D_ + h * DK_ + 2 * p;
  float x0 = in[inOff], x1 = in[inOff + 1];
  if (mode == 1) {
    float inv = __powf(10000.f, -2.f * (float)p / (float)DK_);
    float ang = (float)s * inv;
    float cs = __cosf(ang), sn = __sinf(ang);
    long long o = (long long)((b * H_ + h) * S_ + s) * DK_ + 2 * p;
    out[o]     = x0 * cs - x1 * sn;
    out[o + 1] = x1 * cs + x0 * sn;
  } else {
    long long o = ((long long)(b * H_ + h) * DK_ + 2 * p) * S_ + s;
    out[o]      = x0;
    out[o + S_] = x1;
  }
}

// [B,H,S,DK] -> [B,S,D]
__global__ void unreshape_kernel(const float* __restrict__ in,
                                 float* __restrict__ out) {
  int i = blockIdx.x * blockDim.x + threadIdx.x;
  if (i >= B_ * S_ * D_) return;
  int d = i % DK_; int t = i / DK_;
  int s = t % S_; t /= S_;
  int h = t % H_; int b = t / H_;
  out[(long long)(b * S_ + s) * D_ + h * DK_ + d] = in[i];
}

// in-place masked, scaled softmax over rows of scores [B*H*S, S]
__global__ __launch_bounds__(256) void softmax_mask_kernel(
    float* __restrict__ sc, const int* __restrict__ mask,
    int mask_rows, float scale) {
  int row = blockIdx.x;
  int b = row / (H_ * S_);
  int q = row % S_;
  float* rp = sc + (long long)row * S_;
  const int* mrow = mask + (long long)b * mask_rows * S_
                         + (mask_rows > 1 ? (long long)q * S_ : 0);
  __shared__ float red[256];
  float mx = -3.4e38f;
  for (int j = threadIdx.x; j < S_; j += 256) {
    float v = rp[j] * scale;
    if (mrow[j] == 0) v = -1.0e9f;
    rp[j] = v;
    mx = fmaxf(mx, v);
  }
  red[threadIdx.x] = mx; __syncthreads();
  for (int s = 128; s > 0; s >>= 1) {
    if (threadIdx.x < s) red[threadIdx.x] = fmaxf(red[threadIdx.x], red[threadIdx.x + s]);
    __syncthreads();
  }
  mx = red[0]; __syncthreads();
  float sum = 0.f;
  for (int j = threadIdx.x; j < S_; j += 256) {
    float e = __expf(rp[j] - mx);
    rp[j] = e; sum += e;
  }
  red[threadIdx.x] = sum; __syncthreads();
  for (int s = 128; s > 0; s >>= 1) {
    if (threadIdx.x < s) red[threadIdx.x] += red[threadIdx.x + s];
    __syncthreads();
  }
  float inv = 1.f / red[0];
  for (int j = threadIdx.x; j < S_; j += 256) rp[j] *= inv;
}

// out = alpha * ((x+r) - mean)/sqrt(std_ddof1 + eps) + beta   (D=512, blk=256)
__global__ __launch_bounds__(256) void add_ln_kernel(
    const float* __restrict__ x, const float* __restrict__ r,
    const float* __restrict__ alpha, const float* __restrict__ beta,
    float* __restrict__ out) {
  int row = blockIdx.x;
  const float* xp = x + (long long)row * D_;
  const float* rp = r + (long long)row * D_;
  int t = threadIdx.x;
  float v0 = xp[t] + rp[t];
  float v1 = xp[t + 256] + rp[t + 256];
  __shared__ float red[256];
  red[t] = v0 + v1; __syncthreads();
  for (int s = 128; s > 0; s >>= 1) {
    if (t < s) red[t] += red[t + s];
    __syncthreads();
  }
  float mean = red[0] / (float)D_; __syncthreads();
  float d0 = v0 - mean, d1 = v1 - mean;
  red[t] = d0 * d0 + d1 * d1; __syncthreads();
  for (int s = 128; s > 0; s >>= 1) {
    if (t < s) red[t] += red[t + s];
    __syncthreads();
  }
  float var = red[0] / (float)(D_ - 1);
  float denom = sqrtf(sqrtf(var) + 1e-6f);   // reference: sqrt(std + eps)
  out[(long long)row * D_ + t]       = alpha[t] * d0 / denom + beta[t];
  out[(long long)row * D_ + t + 256] = alpha[t + 256] * d1 / denom + beta[t + 256];
}

// ============================ host side ==============================

struct AttnW {
  const float *wq,*bq,*wk,*bk,*wv,*bv,*wo,*bo;
  unsigned short *twq,*twk,*twv,*two;
};
struct LayerW {
  AttnW attn, xattn;
  const float *n1a,*n1b,*n2a,*n2b,*n3a,*n3b,*ffw1,*ffb1,*ffw2,*ffb2;
  unsigned short *tffw1,*tffw2;
};

extern "C" void kernel_launch(void* const* d_in, const int* in_sizes, int n_in,
                              void* d_out, int out_size, void* d_ws, size_t ws_size,
                              hipStream_t stream) {
  // -------- parse inputs (two possible flatten orders, detected by size) ----
  const float* enc_embed = nullptr; const float* dec_embed = nullptr;
  const float* out_w = nullptr;     const float* out_b = nullptr;
  const int *src = nullptr, *trg = nullptr, *src_mask = nullptr, *tgt_mask = nullptr;
  LayerW enc[NL_], dec[NL_];

  int idx = 0;
  auto nxt = [&]() { return (const float*)d_in[idx++]; };

  bool insertion = (in_sizes[0] == (long long)V_ * D_);
  if (insertion) {
    enc_embed = nxt(); dec_embed = nxt();
    auto rdAttn = [&](AttnW& a) {
      a.wq=nxt(); a.bq=nxt(); a.wk=nxt(); a.bk=nxt();
      a.wv=nxt(); a.bv=nxt(); a.wo=nxt(); a.bo=nxt();
    };
    for (int l = 0; l < NL_; ++l) {
      rdAttn(enc[l].attn);
      enc[l].n1a=nxt(); enc[l].n1b=nxt(); enc[l].n2a=nxt(); enc[l].n2b=nxt();
      enc[l].ffw1=nxt(); enc[l].ffb1=nxt(); enc[l].ffw2=nxt(); enc[l].ffb2=nxt();
    }
    for (int l = 0; l < NL_; ++l) {
      rdAttn(dec[l].attn);
      dec[l].n1a=nxt(); dec[l].n1b=nxt(); dec[l].n2a=nxt(); dec[l].n2b=nxt();
      dec[l].ffw1=nxt(); dec[l].ffb1=nxt(); dec[l].ffw2=nxt(); dec[l].ffb2=nxt();
      rdAttn(dec[l].xattn);
      dec[l].n3a=nxt(); dec[l].n3b=nxt();
    }
    out_w = nxt(); out_b = nxt();
    src      = (const int*)d_in[idx++];
    trg      = (const int*)d_in[idx++];
    src_mask = (const int*)d_in[idx++];
    tgt_mask = (const int*)d_in[idx++];
  } else {
    // jax tree_flatten: dict keys alphabetical at every level
    auto rdAttnS = [&](AttnW& a) {
      a.bk=nxt(); a.bo=nxt(); a.bq=nxt(); a.bv=nxt();
      a.wk=nxt(); a.wo=nxt(); a.wq=nxt(); a.wv=nxt();
    };
    for (int l = 0; l < NL_; ++l) {
      rdAttnS(dec[l].attn);
      dec[l].ffb1=nxt(); dec[l].ffb2=nxt(); dec[l].ffw1=nxt(); dec[l].ffw2=nxt();
      dec[l].n1a=nxt(); dec[l].n1b=nxt(); dec[l].n2a=nxt(); dec[l].n2b=nxt();
      dec[l].n3a=nxt(); dec[l].n3b=nxt();
      rdAttnS(dec[l].xattn);
    }
    dec_embed = nxt();
    for (int l = 0; l < NL_; ++l) {
      rdAttnS(enc[l].attn);
      enc[l].ffb1=nxt(); enc[l].ffb2=nxt(); enc[l].ffw1=nxt(); enc[l].ffw2=nxt();
      enc[l].n1a=nxt(); enc[l].n1b=nxt(); enc[l].n2a=nxt(); enc[l].n2b=nxt();
    }
    enc_embed = nxt();
    out_b = nxt(); out_w = nxt();
    src      = (const int*)d_in[idx++];
    src_mask = (const int*)d_in[idx++];
    tgt_mask = (const int*)d_in[idx++];
    trg      = (const int*)d_in[idx++];
  }

  // -------- workspace layout --------
  const size_t SD = (size_t)MR_ * D_;
  const size_t actFloats = 11 * SD + (size_t)MR_ * DFF_ + (size_t)B_ * H_ * S_ * S_;
  const size_t wtShorts  = (size_t)NL_ * (6 * (size_t)D_ * D_)       // enc attn + ffn
                         + (size_t)NL_ * (10 * (size_t)D_ * D_)      // dec attn+xattn+ffn
                         + (size_t)D_ * V_;                          // out_w
  const bool bf16path = ws_size >= actFloats * 4 + wtShorts * 2 + 256;

  float* W = (float*)d_ws;
  size_t off = 0;
  auto alloc = [&](size_t n) { float* p = W + off; off += n; return p; };
  float* xe  = alloc(SD);
  float* y   = alloc(SD);
  float* q   = alloc(SD);
  float* k   = alloc(SD);
  float* v   = alloc(SD);
  float* qr  = alloc(SD);
  float* kr  = alloc(SD);
  float* vt  = alloc(SD);                          // [B,H,DK,S]
  float* oh  = alloc(SD);                          // [B,H,S,DK]
  float* oc  = alloc(SD);                          // [B,S,D]
  float* tp  = alloc(SD);
  float* ffh = alloc((size_t)MR_ * DFF_);
  float* sc  = alloc((size_t)B_ * H_ * S_ * S_);
  unsigned short* WT = (unsigned short*)(W + off);
  size_t woff = 0;
  auto walloc = [&](size_t n) { unsigned short* p = WT + woff; woff += n; return p; };
  (void)n_in; (void)out_size;

  // -------- bf16 transposed weight mirrors --------
  auto convW = [&](const float* w, unsigned short* t, int K, int N) {
    if (!bf16path) return;
    dim3 g(N / 32, K / 32);
    conv_transpose_kernel<<<g, 256, 0, stream>>>(w, t, K, N);
  };
  auto prepAttn = [&](AttnW& a) {
    a.twq = walloc((size_t)D_ * D_); convW(a.wq, a.twq, D_, D_);
    a.twk = walloc((size_t)D_ * D_); convW(a.wk, a.twk, D_, D_);
    a.twv = walloc((size_t)D_ * D_); convW(a.wv, a.twv, D_, D_);
    a.two = walloc((size_t)D_ * D_); convW(a.wo, a.two, D_, D_);
  };
  for (int l = 0; l < NL_; ++l) {
    prepAttn(enc[l].attn);
    enc[l].tffw1 = walloc((size_t)D_ * DFF_); convW(enc[l].ffw1, enc[l].tffw1, D_, DFF_);
    enc[l].tffw2 = walloc((size_t)DFF_ * D_); convW(enc[l].ffw2, enc[l].tffw2, DFF_, D_);
  }
  for (int l = 0; l < NL_; ++l) {
    prepAttn(dec[l].attn);
    prepAttn(dec[l].xattn);
    dec[l].tffw1 = walloc((size_t)D_ * DFF_); convW(dec[l].ffw1, dec[l].tffw1, D_, DFF_);
    dec[l].tffw2 = walloc((size_t)DFF_ * D_); convW(dec[l].ffw2, dec[l].tffw2, DFF_, D_);
  }
  unsigned short* tout_w = walloc((size_t)D_ * V_); convW(out_w, tout_w, D_, V_);

  // -------- GEMM launch helpers --------
  auto gemmW = [&](const float* A, const float* Wf, const unsigned short* Wt,
                   const float* bias, float* C, int M, int N, int K, bool relu) {
    dim3 g(N / 64, M / 128, 1), blk(128);
    if (bf16path) {
      if (relu) gemm_bf16_wmma<2, true, true ><<<g, blk, 0, stream>>>(A, Wt, bias, C, N, K, 0, 0, 0);
      else      gemm_bf16_wmma<2, true, false><<<g, blk, 0, stream>>>(A, Wt, bias, C, N, K, 0, 0, 0);
    } else {
      if (relu) gemm_bf16_wmma<0, true, true ><<<g, blk, 0, stream>>>(A, Wf, bias, C, N, K, 0, 0, 0);
      else      gemm_bf16_wmma<0, true, false><<<g, blk, 0, stream>>>(A, Wf, bias, C, N, K, 0, 0, 0);
    }
  };
  auto gemmNT = [&](const float* A, const float* Bf, float* C,
                    int M, int N, int K, int batch,
                    long long sa, long long sb, long long scb) {
    dim3 g(N / 64, M / 128, batch), blk(128);
    gemm_bf16_wmma<1, false, false><<<g, blk, 0, stream>>>(A, Bf, nullptr, C, N, K, sa, sb, scb);
  };

  const int rpTotal = B_ * H_ * S_ * (DK_ / 2);
  auto mha = [&](const float* xq, const float* xkv, const AttnW& aw,
                 const int* mask, int mask_rows, float* outp) {
    gemmW(xq,  aw.wq, aw.twq, aw.bq, q, MR_, D_, D_, false);
    gemmW(xkv, aw.wk, aw.twk, aw.bk, k, MR_, D_, D_, false);
    gemmW(xkv, aw.wv, aw.twv, aw.bv, v, MR_, D_, D_, false);
    reshape_rope_kernel<<<(rpTotal + 255) / 256, 256, 0, stream>>>(q, qr, 1);
    reshape_rope_kernel<<<(rpTotal + 255) / 256, 256, 0, stream>>>(k, kr, 1);
    reshape_rope_kernel<<<(rpTotal + 255) / 256, 256, 0, stream>>>(v, vt, 2);
    // scores = qr @ kr^T  (batched over B*H; kr is [S,DK] = NT layout)
    gemmNT(qr, kr, sc, S_, S_, DK_, B_ * H_,
           (long long)S_ * DK_, (long long)S_ * DK_, (long long)S_ * S_);
    softmax_mask_kernel<<<B_ * H_ * S_, 256, 0, stream>>>(sc, mask, mask_rows, 0.125f);
    // context = probs @ v  (vt is [DK,S] per head = NT layout)
    gemmNT(sc, vt, oh, S_, DK_, S_, B_ * H_,
           (long long)S_ * S_, (long long)S_ * DK_, (long long)S_ * DK_);
    unreshape_kernel<<<(int)((SD + 255) / 256), 256, 0, stream>>>(oh, oc);
    gemmW(oc, aw.wo, aw.two, aw.bo, outp, MR_, D_, D_, false);
  };

  auto ffn = [&](const float* xin, const LayerW& L, float* outp) {
    gemmW(xin, L.ffw1, L.tffw1, L.ffb1, ffh, MR_, DFF_, D_, true);
    gemmW(ffh, L.ffw2, L.tffw2, L.ffb2, outp, MR_, D_, DFF_, false);
  };

  auto add_ln = [&](float* xbuf, const float* res, const float* a, const float* b) {
    add_ln_kernel<<<MR_, 256, 0, stream>>>(xbuf, res, a, b, xbuf);
  };

  // -------- encoder --------
  embed_kernel<<<(int)((SD + 255) / 256), 256, 0, stream>>>(enc_embed, src, xe, (int)SD);
  for (int l = 0; l < NL_; ++l) {
    mha(xe, xe, enc[l].attn, src_mask, 1, tp);
    add_ln(xe, tp, enc[l].n1a, enc[l].n1b);
    ffn(xe, enc[l], tp);
    add_ln(xe, tp, enc[l].n2a, enc[l].n2b);
  }

  // -------- decoder --------
  embed_kernel<<<(int)((SD + 255) / 256), 256, 0, stream>>>(dec_embed, trg, y, (int)SD);
  for (int l = 0; l < NL_; ++l) {
    mha(y, y, dec[l].attn, tgt_mask, S_, tp);
    add_ln(y, tp, dec[l].n1a, dec[l].n1b);
    mha(y, xe, dec[l].xattn, src_mask, 1, tp);
    add_ln(y, tp, dec[l].n2a, dec[l].n2b);
    ffn(y, dec[l], tp);
    add_ln(y, tp, dec[l].n3a, dec[l].n3b);
  }

  // -------- vocab projection --------
  gemmW(y, out_w, tout_w, out_b, (float*)d_out, MR_, V_, D_, false);
}
